// Encoder_44238163149250
// MI455X (gfx1250) — matrix operand order repaired and checked by reading, not compiled, not run on hardware
//
#include <hip/hip_runtime.h>

// ---------------- WMMA types & helpers (gfx1250, wave32) ----------------
typedef __attribute__((ext_vector_type(4)))  _Float16 v4h;
typedef __attribute__((ext_vector_type(8)))  _Float16 v8h;
typedef __attribute__((ext_vector_type(16))) _Float16 v16h;
typedef __attribute__((ext_vector_type(4)))  float    v4f;
typedef __attribute__((ext_vector_type(8)))  float    v8f;

__device__ __forceinline__ v8f wmma16(v16h a, v16h b, v8f c) {
    // v_wmma_f32_16x16x32_f16 : D(16x16 f32) = A(16x32 f16) * B(32x16 f16) + C
    return __builtin_amdgcn_wmma_f32_16x16x32_f16(
        /*neg_a=*/false, a, /*neg_b=*/false, b,
        /*c_mod=*/(short)0, c, /*reuse_a=*/false, /*reuse_b=*/false);
}

// A fragment (16x32, f16). ISA layout: lanes 0-15 = rows M=0..15 holding
// K=[ko,ko+8) (VGPR0-3) and K=[ko+16,ko+24) (VGPR4-7); lanes 16-31 hold the
// +8 shifted K chunks for the same rows.  tileIn is [16 rows][kstride] f16.
__device__ __forceinline__ v16h frag_a(const _Float16* tile, int kstride, int ko, int lane) {
    const _Float16* p0 = tile + (lane & 15) * kstride + ko + ((lane >> 4) & 1) * 8;
    v8h lo = *(const v8h*)p0;         // K .. K+7
    v8h hi = *(const v8h*)(p0 + 16);  // K+16 .. K+23
    v16h a;
#pragma unroll
    for (int i = 0; i < 8; ++i) { a[i] = lo[i]; a[i + 8] = hi[i]; }
    return a;
}

// B fragment (32x16, f16) for B = W^T.  Lane l<16 holds column N=l with
// K=[ko,ko+8) and K=[ko+16,ko+24); lanes 16-31 hold the +8 shifted K chunks.
// Row-major W[out][K] -> two 8-half chunks of row n.
__device__ __forceinline__ v16h frag_b(const _Float16* W, int kstride, int nt, int ko, int lane) {
    const _Float16* p0 = W + ((nt << 4) + (lane & 15)) * kstride + ko + ((lane >> 4) & 1) * 8;
    v8h lo = *(const v8h*)p0;
    v8h hi = *(const v8h*)(p0 + 16);
    v16h b;
#pragma unroll
    for (int i = 0; i < 8; ++i) { b[i] = lo[i]; b[i + 8] = hi[i]; }
    return b;
}

// Copy prestaged f16 weights global -> LDS, 8 halves (b128) at a time.
__device__ __forceinline__ void copy_h8(const _Float16* __restrict__ src, _Float16* dst,
                                        int nelem, int tid, int nthr) {
    for (int i = tid; i < (nelem >> 3); i += nthr)
        *(v8h*)(dst + i * 8) = *(const v8h*)(src + i * 8);
}

// One MLP layer (ReLU) for this wave's 16-point tile: LDS f16 -> LDS f16.
template <int KDIM, int ODIM>
__device__ __forceinline__ void layer_lds(const _Float16* tileIn, const _Float16* W,
                                          const float* bias, _Float16* tileOut, int lane) {
    v16h a[KDIM / 32];
#pragma unroll
    for (int ks = 0; ks < KDIM / 32; ++ks) a[ks] = frag_a(tileIn, KDIM, ks * 32, lane);
    const int ro = ((lane >> 4) & 1) * 8;
    const int nl = lane & 15;
#pragma unroll
    for (int nt = 0; nt < ODIM / 16; ++nt) {
        v8f acc = {};
#pragma unroll
        for (int ks = 0; ks < KDIM / 32; ++ks)
            acc = wmma16(a[ks], frag_b(W, KDIM, nt, ks * 32, lane), acc);
        const float bb = bias[nt * 16 + nl];
        _Float16* po = tileOut + ro * ODIM + nt * 16 + nl;
#pragma unroll
        for (int r = 0; r < 8; ++r)
            po[r * ODIM] = (_Float16)fmaxf(acc[r] + bb, 0.0f);
    }
}

// Final MLP layer (no ReLU): LDS f16 -> global f32.  Fast unguarded path when
// the whole 16-row tile is in range (true for all but the tiny agent launch).
template <int KDIM, int ODIM>
__device__ __forceinline__ void layer_out(const _Float16* tileIn, const _Float16* W,
                                          const float* bias, float* out, int prow0,
                                          int npts, int lane) {
    v16h a[KDIM / 32];
#pragma unroll
    for (int ks = 0; ks < KDIM / 32; ++ks) a[ks] = frag_a(tileIn, KDIM, ks * 32, lane);
    const int ro = ((lane >> 4) & 1) * 8;
    const int nl = lane & 15;
    const bool full = (prow0 + 16 <= npts);
    float* orow = out + (size_t)(prow0 + ro) * ODIM + nl;
#pragma unroll
    for (int nt = 0; nt < ODIM / 16; ++nt) {
        v8f acc = {};
#pragma unroll
        for (int ks = 0; ks < KDIM / 32; ++ks)
            acc = wmma16(a[ks], frag_b(W, KDIM, nt, ks * 32, lane), acc);
        const float bb = bias[nt * 16 + nl];
        float* po = orow + nt * 16;
        if (full) {
#pragma unroll
            for (int r = 0; r < 8; ++r) po[r * ODIM] = acc[r] + bb;   // imm-offset stores
        } else {
#pragma unroll
            for (int r = 0; r < 8; ++r)
                if (prow0 + ro + r < npts) po[r * ODIM] = acc[r] + bb;
        }
    }
}

// ---------------- Prep kernels: one-time f32 -> f16 weight images in d_ws ----------------
__global__ __launch_bounds__(256) void k_cvt4(const float* __restrict__ src,
                                              _Float16* __restrict__ dst, int n4) {
    const int i = blockIdx.x * 256 + threadIdx.x;
    if (i < n4) {
        v4f w = *(const v4f*)(src + i * 4);
        v4h h;
        h[0] = (_Float16)w[0]; h[1] = (_Float16)w[1];
        h[2] = (_Float16)w[2]; h[3] = (_Float16)w[3];
        *(v4h*)(dst + i * 4) = h;
    }
}

// Pad K 44 -> 64 while converting (row-major [rows][44] -> [rows][64]).
__global__ __launch_bounds__(256) void k_cvt_pad(const float* __restrict__ src,
                                                 _Float16* __restrict__ dst, int rows) {
    const int i = blockIdx.x * 256 + threadIdx.x;
    if (i < rows * 64) {
        const int j = i >> 6, k = i & 63;
        dst[i] = (k < 44) ? (_Float16)src[j * 44 + k] : (_Float16)0.0f;
    }
}

// ---------------- Kernel 1: dist MLP  (1 -> 128 -> 128) ----------------
// 256 points / block, 256 threads (8 waves x 2 tiles x 16 points).
__global__ __launch_bounds__(256) void k_dist(const float* __restrict__ dist,
                                              const float* __restrict__ W1,
                                              const float* __restrict__ b1,
                                              const _Float16* __restrict__ W2h,
                                              const float* __restrict__ b2,
                                              float* __restrict__ out, int npts) {
    extern __shared__ char smem[];
    _Float16* sW2 = (_Float16*)smem;           // 128*128 f16
    _Float16* sH1 = sW2 + 128 * 128;           // 256*128 f16
    float* sW1 = (float*)(sH1 + 256 * 128);    // 128 f32
    float* sB1 = sW1 + 128;
    float* sB2 = sB1 + 128;
    float* sXP = sB2 + 128;                    // 256 f32: log1p(x_p)

    const int tid = threadIdx.x;
    const int pbase = blockIdx.x * 256;

    copy_h8(W2h, sW2, 128 * 128, tid, 256);
    if (tid < 128) { sW1[tid] = W1[tid]; sB1[tid] = b1[tid]; sB2[tid] = b2[tid]; }
    {   // log1p once per point (all 256 threads)
        int gp = pbase + tid; if (gp >= npts) gp = npts - 1;
        sXP[tid] = log1pf(dist[gp]);
    }
    __builtin_prefetch(dist + pbase + 256 + tid, 0, 1);  // next block's inputs
    __syncthreads();

    // Layer 1 (rank-1): h1[p][j..j+3] = relu(W1[j]*x_p + b1[j]); b64 LDS stores.
    for (int i = tid; i < 256 * 32; i += 256) {
        const int p = i >> 5, j4 = (i & 31) * 4;
        const float x = sXP[p];
        v4h h;
#pragma unroll
        for (int q = 0; q < 4; ++q)
            h[q] = (_Float16)fmaxf(sW1[j4 + q] * x + sB1[j4 + q], 0.0f);
        *(v4h*)(sH1 + p * 128 + j4) = h;
    }
    __syncthreads();

    // Layer 2: 128x128 GEMM on the matrix pipe; 2 tiles per wave.
    const int wave = tid >> 5, lane = tid & 31;
#pragma unroll
    for (int t = 0; t < 2; ++t) {
        const int row0 = wave * 32 + t * 16;
        layer_out<128, 128>(sH1 + row0 * 128, sW2, sB2, out, pbase + row0, npts, lane);
    }
}

// ---------------- Kernel 2: positional-encoding MLP (44->64->128->128) ----------------
// 128 points / block, 256 threads (8 waves x 16 points).  K of layer1 padded 44 -> 64.
__global__ __launch_bounds__(256) void k_feat(const int* __restrict__ idx, int npts,
                                              const _Float16* __restrict__ W1h, const float* __restrict__ b1,
                                              const _Float16* __restrict__ W2h, const float* __restrict__ b2,
                                              const _Float16* __restrict__ W3h, const float* __restrict__ b3,
                                              float t0, float t1,
                                              const int* __restrict__ szr_p, const int* __restrict__ szc_p,
                                              float* __restrict__ out) {
    extern __shared__ char smem[];
    _Float16* sW1 = (_Float16*)smem;           // 64*64   (K padded 44->64)
    _Float16* sW2 = sW1 + 64 * 64;             // 128*64
    _Float16* sW3 = sW2 + 128 * 64;            // 128*128
    _Float16* sX  = sW3 + 128 * 128;           // 128*64  feat (padded)
    _Float16* sH1 = sX + 128 * 64;             // 128*64
    _Float16* sH2 = sH1 + 128 * 64;            // 128*128
    float* sB1 = (float*)(sH2 + 128 * 128);    // 64
    float* sB2 = sB1 + 64;                     // 128
    float* sB3 = sB2 + 128;                    // 128
    float* sCX = sB3 + 128;                    // 128
    float* sCY = sCX + 128;                    // 128

    const int tid = threadIdx.x;
    const int pbase = blockIdx.x * 128;

    // Stage prestaged f16 weights and f32 biases.
    copy_h8(W1h, sW1, 64 * 64, tid, 256);
    copy_h8(W2h, sW2, 128 * 64, tid, 256);
    copy_h8(W3h, sW3, 128 * 128, tid, 256);
    if (tid < 64) sB1[tid] = b1[tid];
    if (tid < 128) { sB2[tid] = b2[tid]; sB3[tid] = b3[tid]; }

    // Stage normalized coords.
    if (tid < 128) {
        int gp = pbase + tid; if (gp >= npts) gp = npts - 1;
        sCX[tid] = (float)idx[gp * 2 + 0] / (float)szr_p[0];
        sCY[tid] = (float)idx[gp * 2 + 1] / (float)szc_p[0];
    }
    __syncthreads();

    // Positional encoding: [x, y, {sin(fx),sin(fy),cos(fx),cos(fy)}*10, t0, t1, 0-pad]
    for (int i = tid; i < 128 * 64; i += 256) {
        const int p = i >> 6, k = i & 63;
        const float cx = sCX[p], cy = sCY[p];
        float v;
        if (k == 0)      v = cx;
        else if (k == 1) v = cy;
        else if (k < 42) {
            const int q = k - 2, f = q >> 2, r = q & 3;
            const float arg = (float)(1 << f) * ((r & 1) ? cy : cx);
            v = (r < 2) ? __sinf(arg) : __cosf(arg);
        }
        else if (k == 42) v = t0;
        else if (k == 43) v = t1;
        else              v = 0.0f;
        sX[i] = (_Float16)v;
    }
    __syncthreads();

    const int wave = tid >> 5, lane = tid & 31;
    layer_lds<64, 64>(sX + wave * 16 * 64, sW1, sB1, sH1 + wave * 16 * 64, lane);
    __syncthreads();
    layer_lds<64, 128>(sH1 + wave * 16 * 64, sW2, sB2, sH2 + wave * 16 * 128, lane);
    __syncthreads();
    layer_out<128, 128>(sH2 + wave * 16 * 128, sW3, sB3, out, pbase + wave * 16, npts, lane);
}

// ---------------- Host launcher ----------------
extern "C" void kernel_launch(void* const* d_in, const int* in_sizes, int n_in,
                              void* d_out, int out_size, void* d_ws, size_t ws_size,
                              hipStream_t stream) {
    (void)n_in; (void)out_size; (void)ws_size;
    const int* frontier = (const int*)d_in[0];
    const int* agent    = (const int*)d_in[1];
    const int* phist    = (const int*)d_in[2];
    const int* ghist    = (const int*)d_in[3];
    const float* dist   = (const float*)d_in[4];
    const float *Wf1 = (const float*)d_in[5],  *bf1 = (const float*)d_in[6];
    const float *Wf2 = (const float*)d_in[7],  *bf2 = (const float*)d_in[8];
    const float *Wf3 = (const float*)d_in[9],  *bf3 = (const float*)d_in[10];
    const float *Wa1 = (const float*)d_in[11], *ba1 = (const float*)d_in[12];
    const float *Wa2 = (const float*)d_in[13], *ba2 = (const float*)d_in[14];
    const float *Wa3 = (const float*)d_in[15], *ba3 = (const float*)d_in[16];
    const float *Wd1 = (const float*)d_in[17], *bd1 = (const float*)d_in[18];
    const float *Wd2 = (const float*)d_in[19], *bd2 = (const float*)d_in[20];
    const int* szr = (const int*)d_in[21];
    const int* szc = (const int*)d_in[22];
    float* out = (float*)d_out;

    const int nDist  = in_sizes[4];        // B*D*NF = 524288
    const int nFront = in_sizes[0] / 2;    // 65536
    const int nAgent = in_sizes[1] / 2;    // 64
    const int nPh    = in_sizes[2] / 2;    // 16384
    const int nGh    = in_sizes[3] / 2;    // 16384

    float* out_dist  = out;
    float* out_front = out_dist  + (size_t)nDist  * 128;
    float* out_agent = out_front + (size_t)nFront * 128;
    float* out_ph    = out_agent + (size_t)nAgent * 128;
    float* out_gh    = out_ph    + (size_t)nPh    * 128;

    // f16 weight images in scratch (147456 bytes total).
    _Float16* ws   = (_Float16*)d_ws;
    _Float16* Wd2h = ws;                 // 16384
    _Float16* Wf1h = Wd2h + 16384;       // 4096 (padded 64x64)
    _Float16* Wf2h = Wf1h + 4096;        // 8192
    _Float16* Wf3h = Wf2h + 8192;        // 16384
    _Float16* Wa1h = Wf3h + 16384;       // 4096 (padded 64x64)
    _Float16* Wa2h = Wa1h + 4096;        // 8192
    _Float16* Wa3h = Wa2h + 8192;        // 16384

    k_cvt4<<<16, 256, 0, stream>>>(Wd2, Wd2h, 4096);
    k_cvt_pad<<<16, 256, 0, stream>>>(Wf1, Wf1h, 64);
    k_cvt4<<<8, 256, 0, stream>>>(Wf2, Wf2h, 2048);
    k_cvt4<<<16, 256, 0, stream>>>(Wf3, Wf3h, 4096);
    k_cvt_pad<<<16, 256, 0, stream>>>(Wa1, Wa1h, 64);
    k_cvt4<<<8, 256, 0, stream>>>(Wa2, Wa2h, 2048);
    k_cvt4<<<16, 256, 0, stream>>>(Wa3, Wa3h, 4096);

    const size_t lds_dist = (size_t)(128 * 128 + 256 * 128) * 2 + (size_t)(3 * 128 + 256) * 4;
    const size_t lds_feat = (size_t)(64 * 64 + 128 * 64 + 128 * 128 +
                                     128 * 64 + 128 * 64 + 128 * 128) * 2 +
                            (size_t)(64 + 128 + 128 + 128 + 128) * 4;

    k_dist<<<(nDist + 255) / 256, 256, lds_dist, stream>>>(dist, Wd1, bd1, Wd2h, bd2,
                                                           out_dist, nDist);
    k_feat<<<(nFront + 127) / 128, 256, lds_feat, stream>>>(frontier, nFront,
        Wf1h, bf1, Wf2h, bf2, Wf3h, bf3, 1.0f, 0.0f, szr, szc, out_front);
    k_feat<<<(nAgent + 127) / 128, 256, lds_feat, stream>>>(agent, nAgent,
        Wa1h, ba1, Wa2h, ba2, Wa3h, ba3, 0.0f, 1.0f, szr, szc, out_agent);
    k_feat<<<(nPh + 127) / 128, 256, lds_feat, stream>>>(phist, nPh,
        Wa1h, ba1, Wa2h, ba2, Wa3h, ba3, 0.0f, 1.0f, szr, szc, out_ph);
    k_feat<<<(nGh + 127) / 128, 256, lds_feat, stream>>>(ghist, nGh,
        Wf1h, bf1, Wf2h, bf2, Wf3h, bf3, 1.0f, 0.0f, szr, szc, out_gh);
}
